// RNNModel_75591424409846
// MI455X (gfx1250) — matrix operand order repaired and checked by reading, not compiled, not run on hardware
//
#include <hip/hip_runtime.h>

typedef __attribute__((ext_vector_type(16))) _Float16 v16h;
typedef __attribute__((ext_vector_type(8)))  float    v8f;

#define T_STEPS 2048
#define BATCH   256
#define HID     64
#define LDSROW  72   // padded h row stride in halves: 144B = 36 dwords, 36*m mod 64 distinct

// Native v_tanh_f32 on gfx1250 (confirmed in round-3 disasm); exp/rcp fallback.
__device__ __forceinline__ float fast_tanh(float v) {
#if __has_builtin(__builtin_amdgcn_tanhf)
    return __builtin_amdgcn_tanhf(v);
#else
    float e = __builtin_amdgcn_exp2f(v * 2.88539008177792681472f); // 2*log2(e)
    float r = __builtin_amdgcn_rcpf(e + 1.0f);
    return __builtin_fmaf(-2.0f, r, 1.0f);
#endif
}

// Workgroup barrier that publishes only LDS traffic (no storecnt drain):
// out-stores keep pipelining across steps.
__device__ __forceinline__ void lds_barrier() {
    asm volatile(
        "s_wait_dscnt 0x0\n\t"
        "s_barrier_signal -1\n\t"
        "s_barrier_wait -1"
        ::: "memory");
}

// One recurrence step. xv holds x[t] for this lane's 8 rows (register-pipelined);
// issues the loads for x[t+1] into *xnext early so their latency hides under the
// WMMA/tanh/barrier work of this step.
__device__ __forceinline__ void rnn_step(
    const _Float16* __restrict__ hin, _Float16* __restrict__ hout,
    const v8f xv, v8f* __restrict__ xnext,
    int t, const float* __restrict__ x_glob,
    const v16h& b0, const v16h& b1,
    float bias_n, float wih_n,
    int mn, int grp, int jcol, int brow0,
    float* __restrict__ out, float d[8])
{
    // ---- A fragments (16-bit A layout, ISA 7.12.2): row m = mn;
    // elem pair (2v,2v+1) -> K = (v<4?0:16) + grp*8 + (v&3)*2 + {0,1}
    v16h a0, a1;
#pragma unroll
    for (int v = 0; v < 8; ++v) {
        int k = ((v < 4) ? 0 : 16) + grp * 8 + (v & 3) * 2;
        a0[2 * v]     = hin[mn * LDSROW + k];
        a0[2 * v + 1] = hin[mn * LDSROW + k + 1];
        a1[2 * v]     = hin[mn * LDSROW + 32 + k];
        a1[2 * v + 1] = hin[mn * LDSROW + 32 + k + 1];
    }

    // ---- prefetch x for step t+1 (clamped; values consumed next step) ----
    {
        int tn = (t + 1 < T_STEPS) ? (t + 1) : (T_STEPS - 1);
        *xnext = *(const v8f*)(x_glob + (size_t)tn * BATCH + brow0 + grp * 8);
    }

    // ---- C init: bias + x_t * W_ih (rank-1, IN==1). D layout: VGPR v -> m = v+grp*8
    v8f c;
#pragma unroll
    for (int v = 0; v < 8; ++v)
        c[v] = __builtin_fmaf(xv[v], wih_n, bias_n);

    c = __builtin_amdgcn_wmma_f32_16x16x32_f16(false, a0, false, b0,
                                               (short)0, c, false, false);
    c = __builtin_amdgcn_wmma_f32_16x16x32_f16(false, a1, false, b1,
                                               (short)0, c, false, false);

#pragma unroll
    for (int v = 0; v < 8; ++v)
        d[v] = fast_tanh(c[v]);

    // ---- stream out[t, brow0+m, jcol] (fire-and-forget; no fence at barrier) ----
    size_t obase = (size_t)t * (BATCH * HID);
#pragma unroll
    for (int v = 0; v < 8; ++v)
        out[obase + (size_t)(brow0 + v + grp * 8) * HID + jcol] = d[v];

    // ---- write h_new (f16) into the *other* buffer ----
#pragma unroll
    for (int v = 0; v < 8; ++v)
        hout[(v + grp * 8) * LDSROW + jcol] = (_Float16)d[v];

    lds_barrier();   // publish hout (dscnt only), retire hin reads
}

__global__ __launch_bounds__(128)
void rnn_scan_kernel(const float* __restrict__ x,
                     const float* __restrict__ hidden,
                     const float* __restrict__ W_ih,
                     const float* __restrict__ W_hh,
                     const float* __restrict__ b_ih,
                     const float* __restrict__ b_hh,
                     float* __restrict__ out,       // [T,B,H]
                     float* __restrict__ h_final)   // [B,H]
{
    __shared__ _Float16 hbuf0[16 * LDSROW];
    __shared__ _Float16 hbuf1[16 * LDSROW];

    const int tid   = threadIdx.x;
    const int lane  = tid & 31;
    const int wave  = tid >> 5;
    const int grp   = lane >> 4;     // 0: lanes 0-15, 1: lanes 16-31
    const int mn    = lane & 15;     // A: row m;  B/C/D: column n (within tile)
    const int brow0 = blockIdx.x * 16;

    const int jcol = wave * 16 + mn; // global hidden column owned by this lane

    // ---- stationary B fragments: B[k][n] = W_hh[jcol*HID + k] ----
    // 16-bit B layout: lane = column n; K = (e<8 ? 0 : 16) + grp*8 + (e&7) (per 32-wide K-half)
    v16h b0, b1;
#pragma unroll
    for (int e = 0; e < 16; ++e) {
        int k = ((e < 8) ? 0 : 16) + grp * 8 + (e & 7);
        b0[e] = (_Float16)W_hh[jcol * HID + k];        // K = 0..31
        b1[e] = (_Float16)W_hh[jcol * HID + 32 + k];   // K = 32..63
    }

    const float bias_n = b_ih[jcol] + b_hh[jcol];
    const float wih_n  = W_ih[jcol];                   // IN == 1

    // ---- init: h0 tile (padded rows) + x[0] into registers ----
    for (int i = tid; i < 16 * HID; i += 128) {
        int r = i >> 6, cidx = i & 63;
        hbuf0[r * LDSROW + cidx] = (_Float16)hidden[brow0 * HID + i];
    }
    v8f xv = *(const v8f*)(x + brow0 + grp * 8);   // x[0] for this lane's 8 rows
    v8f xn;
    __syncthreads();

    float d[8];
#pragma unroll 1
    for (int t = 0; t < T_STEPS; t += 2) {
        rnn_step(hbuf0, hbuf1, xv, &xn, t,     x, b0, b1, bias_n, wih_n,
                 mn, grp, jcol, brow0, out, d);
        rnn_step(hbuf1, hbuf0, xn, &xv, t + 1, x, b0, b1, bias_n, wih_n,
                 mn, grp, jcol, brow0, out, d);
    }

    // d holds h at t = T-1 (full f32 precision)
#pragma unroll
    for (int v = 0; v < 8; ++v)
        h_final[(size_t)(brow0 + v + grp * 8) * HID + jcol] = d[v];
}

extern "C" void kernel_launch(void* const* d_in, const int* in_sizes, int n_in,
                              void* d_out, int out_size, void* d_ws, size_t ws_size,
                              hipStream_t stream) {
    const float* x      = (const float*)d_in[0];  // [T,B,IN]
    const float* hidden = (const float*)d_in[1];  // [B,H]
    const float* W_ih   = (const float*)d_in[2];  // [H,IN]
    const float* W_hh   = (const float*)d_in[3];  // [H,H]
    const float* b_ih   = (const float*)d_in[4];  // [H]
    const float* b_hh   = (const float*)d_in[5];  // [H]

    float* out     = (float*)d_out;                        // [T,B,H]
    float* h_final = out + (size_t)T_STEPS * BATCH * HID;  // [B,H]

    rnn_scan_kernel<<<BATCH / 16, 128, 0, stream>>>(
        x, hidden, W_ih, W_hh, b_ih, b_hh, out, h_final);
}
